// STGCNModel_38637525795273
// MI455X (gfx1250) — matrix operand order repaired and checked by reading, not compiled, not run on hardware
//
#include <hip/hip_runtime.h>

typedef __attribute__((ext_vector_type(16))) _Float16 v16h;
typedef __attribute__((ext_vector_type(8)))  _Float16 v8h;
typedef __attribute__((ext_vector_type(4)))  _Float16 v4h;
typedef __attribute__((ext_vector_type(8)))  float    v8f;
typedef _Float16 f16;

#define NB   8
#define CIN0 2
#define TT   300
#define VV   75
#define NTT  (NB*TT)          // 2400
#define PP   (NB*TT*VV)       // 180000
#define NTV  (TT*VV)          // 22500 positions per sample
#define TP8  ((TT+8)*VV)      // 23100 padded rows per sample
#define NSUB (PP/16)          // 11250 position subtiles
#define EPSB 1e-5f

// ---------------- data_bn stats (per c*V+v channel, over N,T) ----------------
__global__ void bn0_stats_k(const float* __restrict__ x, const float* __restrict__ g,
                            const float* __restrict__ b, float* __restrict__ sc,
                            float* __restrict__ sh) {
    int ch = blockIdx.x;                   // ch = c*VV + v
    int c = ch / VV, v = ch % VV;
    __shared__ float ssum[256], ssq[256];
    float s = 0.f, q = 0.f;
    for (int i = threadIdx.x; i < NTT; i += 256) {
        int n = i / TT, t = i % TT;
        float val = x[((n*CIN0 + c)*TT + t)*VV + v];
        s += val; q += val*val;
    }
    ssum[threadIdx.x] = s; ssq[threadIdx.x] = q;
    __syncthreads();
    for (int st = 128; st > 0; st >>= 1) {
        if (threadIdx.x < st) {
            ssum[threadIdx.x] += ssum[threadIdx.x+st];
            ssq [threadIdx.x] += ssq [threadIdx.x+st];
        }
        __syncthreads();
    }
    if (threadIdx.x == 0) {
        float m   = ssum[0] / (float)NTT;
        float var = ssq[0]  / (float)NTT - m*m;
        float scl = g[ch] * rsqrtf(var + EPSB);
        sc[ch] = scl;
        sh[ch] = b[ch] - m*scl;
    }
}

// apply data_bn -> channels-last f16 [P][32], channels 2..31 zeroed (K padding)
__global__ void bn0_apply_k(const float* __restrict__ x, const float* __restrict__ sc,
                            const float* __restrict__ sh, f16* __restrict__ h) {
    int i = blockIdx.x*256 + threadIdx.x;   // over PP*32
    if (i >= PP*32) return;
    int c = i & 31, p = i >> 5;
    float val = 0.f;
    if (c < CIN0) {
        int n = p / NTV; int r = p % NTV; int t = r / VV; int v = r % VV;
        int ch = c*VV + v;
        val = x[((n*CIN0 + c)*TT + t)*VV + v]*sc[ch] + sh[ch];
    }
    h[i] = (f16)val;
}

// ------- spatial agg, channels-last: y[(nt,w)*S+c] = sum_v x[(nt,v)*S+c]*A[v][w]
__global__ void agg_k(const f16* __restrict__ xh, f16* __restrict__ yh,
                      const float* __restrict__ A, int ci, int S) {
    __shared__ float At[VV][VV + 1];        // At[w][v] = A[v][w]
    for (int i = threadIdx.x; i < VV*VV; i += 256) {
        int v = i / VV, w = i % VV;
        At[w][v] = A[i];
    }
    __syncthreads();
    int total = NTT * VV * S;
    for (int i = blockIdx.x*256 + threadIdx.x; i < total; i += gridDim.x*256) {
        int c  = i % S;
        int pw = i / S;
        int w  = pw % VV;
        int nt = pw / VV;
        float acc = 0.f;
        if (c < ci) {
            const f16* xr = xh + (size_t)nt*VV*S + c;
            #pragma unroll 5
            for (int v = 0; v < VV; ++v) acc += (float)xr[(size_t)v*S] * At[w][v];
        }
        yh[i] = (f16)acc;
    }
}

// ---------------- weight converts ----------------
__global__ void f32_to_f16_k(const float* __restrict__ s, f16* __restrict__ d, int n) {
    int i = blockIdx.x*256 + threadIdx.x;
    if (i < n) d[i] = (f16)s[i];
}
// tcn_w OIHW [M][C][9][1] f32  ->  tap-major f16 [9][M][C]
__global__ void tcnw_cvt_k(const float* __restrict__ s, f16* __restrict__ d, int M, int C) {
    int i = blockIdx.x*256 + threadIdx.x;
    if (i >= M*C*9) return;
    int o = i / (C*9); int r = i % (C*9); int c = r / 9; int k = r % 9;
    d[((size_t)k*M + o)*C + c] = (f16)s[i];
}

// ============ channels-last WMMA GEMM / implicit temporal conv ============
// Y[p][o] = sum_{tap,k} W[tap][o][k] * X[row(p)+dt(tap)*VV][k]
// X rows have stride Kloop halves (zero-padded channels beyond Ktrue).
// padded!=0: X rows are the T+8 zero-padded layout; dt offsets are free.
// Workgroup: 128 threads / 4 waves; 32 position-subtiles (512 rows) x 16 couts.
__global__ __launch_bounds__(128) void gemmCL_k(const f16* __restrict__ W,
                                                const f16* __restrict__ X,
                                                float* __restrict__ Y,
                                                int M, int Ktrue, int Kloop,
                                                int ntaps, int padded) {
    __shared__ f16 lWB[16][40];             // W^T tile: lWB[o][kk]
    const int tid = threadIdx.x, lane = tid & 31, wv = tid >> 5;
    const int hi = lane >> 4, lo = lane & 15;
    const int o0 = blockIdx.y * 16;
    const int sbase = blockIdx.x * 32;
    const bool kfull = (Ktrue & 31) == 0;   // uniform: no K zero-fill needed

    int rowidx[8];                          // per-lane input row for each subtile
    #pragma unroll
    for (int j = 0; j < 8; ++j) {
        int s = sbase + j*4 + wv;
        int p = s*16 + lo;
        if (p >= PP) p = PP - 1;            // clamp; result discarded at store
        int row = p;
        if (padded) {
            int n = p / NTV;
            int local = p - n*NTV;
            row = n*TP8 + 4*VV + local;
        }
        rowidx[j] = row;
    }
    v8f acc[8] = {};

    for (int tap = 0; tap < ntaps; ++tap) {
        const int dt = (ntaps == 9) ? (tap - 4) : 0;
        const f16* Wt = W + (size_t)tap * M * Ktrue;
        for (int c0 = 0; c0 < Kloop; c0 += 32) {
            // stage 16x32 weight tile (4 halves per thread)
            {
                int o = tid >> 3, kk = (tid & 7) * 4;
                const f16* wsrc = Wt + (size_t)(o0 + o)*Ktrue + c0 + kk;
                if (kfull) {
                    *(v4h*)&lWB[o][kk] = *(const v4h*)wsrc;   // one b64 load+store
                } else {
                    #pragma unroll
                    for (int u = 0; u < 4; ++u) {
                        int k = c0 + kk + u;
                        lWB[o][kk + u] = (k < Ktrue) ? wsrc[u] : (f16)0.0f;
                    }
                }
                if (c0 + 32 < Kloop)        // warm next weight chunk
                    __builtin_prefetch(Wt + (size_t)(o0 + o)*Ktrue + c0 + 32, 0, 0);
            }
            __syncthreads();
            // B fragment: K = 8*hi + (e&7) + 16*(e>>3), N = lo
            v16h bf;
            {
                const f16* brow = &lWB[lo][8*hi];
                #pragma unroll
                for (int e = 0; e < 16; ++e) bf[e] = brow[(e & 7) + (e >> 3)*16];
            }
            // 8 subtiles share this B fragment
            #pragma unroll
            for (int j = 0; j < 8; ++j) {
                const f16* xr = X + (size_t)(rowidx[j] + dt*VV)*Kloop + c0;
                v8h a0 = *(const v8h*)(xr + 8*hi);        // K = c0+8hi .. +7
                v8h a1 = *(const v8h*)(xr + 16 + 8*hi);   // K = c0+16+8hi .. +7
                v16h a = __builtin_shufflevector(a0, a1, 0,1,2,3,4,5,6,7,
                                                 8,9,10,11,12,13,14,15);
                acc[j] = __builtin_amdgcn_wmma_f32_16x16x32_f16(
                    false, a, false, bf, (short)0, acc[j], false, false);
            }
            __syncthreads();
        }
    }
    // D layout: position = s*16 + r + 8*hi, cout = o0 + lo
    #pragma unroll
    for (int j = 0; j < 8; ++j) {
        int s = sbase + j*4 + wv;
        if (s*16 >= PP) continue;           // uniform per wave
        #pragma unroll
        for (int r = 0; r < 8; ++r)
            Y[(size_t)(s*16 + r + 8*hi)*M + o0 + lo] = acc[j][r];
    }
}

// ---------------- BN over channels-last [P][co] ----------------
__global__ void zero_f32_k(float* __restrict__ p, int n) {
    int i = blockIdx.x*256 + threadIdx.x;
    if (i < n) p[i] = 0.f;
}
__global__ void bn_partial_k(const float* __restrict__ Y, float* __restrict__ bsum,
                             float* __restrict__ bsq, int co) {
    size_t total = (size_t)PP * co;
    size_t start = (size_t)blockIdx.x*256 + threadIdx.x;
    int c = (int)(start % co);              // stride 512*256 is a multiple of co
    float s = 0.f, q = 0.f;
    for (size_t i = start; i < total; i += (size_t)512*256) {
        float v = Y[i]; s += v; q += v*v;
    }
    atomicAdd(&bsum[c], s);
    atomicAdd(&bsq[c], q);
}
__global__ void bn_final_k(const float* __restrict__ bsum, const float* __restrict__ bsq,
                           const float* __restrict__ g, const float* __restrict__ b,
                           float* __restrict__ sc, float* __restrict__ sh, int co) {
    int c = threadIdx.x;
    if (c >= co) return;
    float m   = bsum[c] / (float)PP;
    float var = bsq[c]  / (float)PP - m*m;
    float scl = g[c] * rsqrtf(var + EPSB);
    sc[c] = scl;
    sh[c] = b[c] - m*scl;
}

// zero the 8 pad time-rows per sample of the padded TCN input
__global__ void padzero_k(f16* __restrict__ HT, int co) {
    int i = blockIdx.x*256 + threadIdx.x;   // over NB*8*VV*co
    if (i >= NB*8*VV*co) return;
    int c = i % co; int r = i / co;
    int v = r % VV; r /= VV;
    int which = r % 8; int n = r / 8;
    int tpad = (which < 4) ? which : (300 + which);   // 0..3, 304..307
    HT[(size_t)(n*TP8 + tpad*VV + v)*co + c] = (f16)0.0f;
}
// BN-normalize gcn output into the zero-padded channels-last TCN input
__global__ void bn_apply_pad_k(const float* __restrict__ Y, const float* __restrict__ sc,
                               const float* __restrict__ sh, f16* __restrict__ HT, int co) {
    size_t i = (size_t)blockIdx.x*256 + threadIdx.x;
    if (i >= (size_t)PP*co) return;
    int c = (int)(i % co);
    int p = (int)(i / co);
    int n = p / NTV; int local = p - n*NTV;
    size_t row = (size_t)n*TP8 + 4*VV + local;
    HT[row*co + c] = (f16)(Y[i]*sc[c] + sh[c]);
}

// out = relu( relu(bn(tcn)) + bn(res) ), channels-last f16
__global__ void combine_k(const float* __restrict__ y2, const float* __restrict__ r,
                          const float* __restrict__ s2, const float* __restrict__ h2,
                          const float* __restrict__ s3, const float* __restrict__ h3,
                          f16* __restrict__ out, int co) {
    size_t i = (size_t)blockIdx.x*256 + threadIdx.x;
    if (i >= (size_t)PP*co) return;
    int c = (int)(i % co);
    float a = fmaxf(y2[i]*s2[c] + h2[c], 0.f);
    float b = r[i]*s3[c] + h3[c];
    out[i] = (f16)fmaxf(a + b, 0.f);
}

// ---------------- pool (channels-last) + FC ----------------
__global__ void pool_partial_k(const f16* __restrict__ h, float* __restrict__ pooled) {
    int n = blockIdx.x >> 5, slice = blockIdx.x & 31;
    int c = threadIdx.x;                    // 256 channels
    float s = 0.f;
    for (int j = slice; j < NTV; j += 32)
        s += (float)h[((size_t)n*NTV + j)*256 + c];
    atomicAdd(&pooled[n*256 + c], s);
}
__global__ void fc_k(const float* __restrict__ pooled, const float* __restrict__ w,
                     const float* __restrict__ b, float* __restrict__ out) {
    int i = blockIdx.x*256 + threadIdx.x;   // 8*52 outputs
    if (i >= NB*52) return;
    int n = i / 52, cls = i % 52;
    float s = b[cls];
    const float inv = 1.0f / (float)NTV;
    for (int c = 0; c < 256; ++c) s += pooled[n*256 + c] * inv * w[cls*256 + c];
    out[i] = s;
}

// ---------------- host ----------------
extern "C" void kernel_launch(void* const* d_in, const int* in_sizes, int n_in,
                              void* d_out, int out_size, void* d_ws, size_t ws_size,
                              hipStream_t stream) {
    (void)in_sizes; (void)n_in; (void)out_size; (void)ws_size;
    const float* x     = (const float*)d_in[0];
    const float* A     = (const float*)d_in[1];
    const float* bn0_g = (const float*)d_in[2];
    const float* bn0_b = (const float*)d_in[3];
    const float* fc_w  = (const float*)d_in[40];
    const float* fc_b  = (const float*)d_in[41];

    char* ws = (char*)d_ws;
    size_t off = 0;
    auto take = [&](size_t bytes) { char* p = ws + off; off += (bytes + 255) & ~(size_t)255; return p; };
    f16*   HA  = (f16*)  take((size_t)256*PP*sizeof(f16));    // activation ping (channels-last)
    f16*   HB  = (f16*)  take((size_t)256*PP*sizeof(f16));    // activation pong
    f16*   HAG = (f16*)  take((size_t)128*PP*sizeof(f16));    // agg output
    f16*   HT  = (f16*)  take((size_t)256*NB*TP8*sizeof(f16)); // padded TCN input
    float* F1  = (float*)take((size_t)256*PP*sizeof(float));  // gcn/tcn out [P][co]
    float* F2  = (float*)take((size_t)256*PP*sizeof(float));  // residual out
    f16*   WG  = (f16*)  take(256*128*sizeof(f16));
    f16*   WR  = (f16*)  take(256*128*sizeof(f16));
    f16*   WT  = (f16*)  take((size_t)9*256*256*sizeof(f16));
    float* S0  = (float*)take(150*sizeof(float));
    float* T0  = (float*)take(150*sizeof(float));
    float* S1  = (float*)take(256*sizeof(float));
    float* H1  = (float*)take(256*sizeof(float));
    float* S2  = (float*)take(256*sizeof(float));
    float* H2  = (float*)take(256*sizeof(float));
    float* S3  = (float*)take(256*sizeof(float));
    float* H3  = (float*)take(256*sizeof(float));
    float* BSUM= (float*)take(256*sizeof(float));
    float* BSQ = (float*)take(256*sizeof(float));
    float* PO  = (float*)take(NB*256*sizeof(float));

    // data_bn -> channels-last f16 [P][32] (channels >=2 zeroed)
    bn0_stats_k<<<CIN0*VV, 256, 0, stream>>>(x, bn0_g, bn0_b, S0, T0);
    bn0_apply_k<<<(PP*32 + 255)/256, 256, 0, stream>>>(x, S0, T0, HA);

    const int cins[3]  = {CIN0, 64, 128};
    const int couts[3] = {64, 128, 256};
    f16* hin = HA; f16* hout = HB;

    for (int blk = 0; blk < 3; ++blk) {
        int base = 4 + 12*blk;
        const float* gcn_w  = (const float*)d_in[base+0];
        const float* gcn_g  = (const float*)d_in[base+2];
        const float* gcn_bb = (const float*)d_in[base+3];
        const float* tcn_w  = (const float*)d_in[base+4];
        const float* tcn_g  = (const float*)d_in[base+6];
        const float* tcn_bb = (const float*)d_in[base+7];
        const float* res_w  = (const float*)d_in[base+8];
        const float* res_g  = (const float*)d_in[base+10];
        const float* res_bb = (const float*)d_in[base+11];
        // gcn_b/tcn_b/res_b cancel exactly in training-mode BN.
        int ci = cins[blk], co = couts[blk];
        int S = (ci < 32) ? 32 : ci;        // zero-padded channel stride

        f32_to_f16_k<<<(co*ci + 255)/256, 256, 0, stream>>>(gcn_w, WG, co*ci);
        f32_to_f16_k<<<(co*ci + 255)/256, 256, 0, stream>>>(res_w, WR, co*ci);
        tcnw_cvt_k<<<(co*co*9 + 255)/256, 256, 0, stream>>>(tcn_w, WT, co, co);

        int na = NTT*VV*S;
        agg_k<<<(na + 255)/256, 256, 0, stream>>>(hin, HAG, A, ci, S);

        dim3 g1((NSUB + 31)/32, co/16);
        // gcn 1x1: F1[P][co] = W x HAG
        gemmCL_k<<<g1, 128, 0, stream>>>(WG, HAG, F1, co, ci, S, 1, 0);
        zero_f32_k<<<2, 256, 0, stream>>>(BSUM, 256);
        zero_f32_k<<<2, 256, 0, stream>>>(BSQ, 256);
        bn_partial_k<<<512, 256, 0, stream>>>(F1, BSUM, BSQ, co);
        bn_final_k<<<1, 256, 0, stream>>>(BSUM, BSQ, gcn_g, gcn_bb, S1, H1, co);

        padzero_k<<<(NB*8*VV*co + 255)/256, 256, 0, stream>>>(HT, co);
        {
            size_t n = (size_t)PP*co;
            bn_apply_pad_k<<<(unsigned)((n + 255)/256), 256, 0, stream>>>(F1, S1, H1, HT, co);
        }
        // TCN implicit GEMM over 9 temporal taps on padded input
        gemmCL_k<<<g1, 128, 0, stream>>>(WT, HT, F1, co, co, co, 9, 1);
        zero_f32_k<<<2, 256, 0, stream>>>(BSUM, 256);
        zero_f32_k<<<2, 256, 0, stream>>>(BSQ, 256);
        bn_partial_k<<<512, 256, 0, stream>>>(F1, BSUM, BSQ, co);
        bn_final_k<<<1, 256, 0, stream>>>(BSUM, BSQ, tcn_g, tcn_bb, S2, H2, co);

        // residual 1x1 on block input
        gemmCL_k<<<g1, 128, 0, stream>>>(WR, hin, F2, co, ci, S, 1, 0);
        zero_f32_k<<<2, 256, 0, stream>>>(BSUM, 256);
        zero_f32_k<<<2, 256, 0, stream>>>(BSQ, 256);
        bn_partial_k<<<512, 256, 0, stream>>>(F2, BSUM, BSQ, co);
        bn_final_k<<<1, 256, 0, stream>>>(BSUM, BSQ, res_g, res_bb, S3, H3, co);

        {
            size_t n = (size_t)PP*co;
            combine_k<<<(unsigned)((n + 255)/256), 256, 0, stream>>>(F1, F2, S2, H2, S3, H3, hout, co);
        }
        f16* tmp = hin; hin = hout; hout = tmp;
    }

    zero_f32_k<<<(NB*256 + 255)/256, 256, 0, stream>>>(PO, NB*256);
    pool_partial_k<<<NB*32, 256, 0, stream>>>(hin, PO);
    fc_k<<<(NB*52 + 255)/256, 256, 0, stream>>>(PO, fc_w, fc_b, (float*)d_out);
}
